// Cogment_89489938580108
// MI455X (gfx1250) — matrix operand order repaired and checked by reading, not compiled, not run on hardware
//
#include <hip/hip_runtime.h>
#include <hip/hip_bf16.h>
#include <math.h>

typedef __attribute__((ext_vector_type(16))) _Float16 v16h;
typedef __attribute__((ext_vector_type(8)))  float    v8f;
typedef __attribute__((ext_vector_type(4)))  float    v4f;

#define SEQ   256
#define DIM   512
#define NHEAD 8
#define HD    64
#define FF    1536
#define RCF   2048
#define VOC   32000
#define NL    2
#define NSTEP 4

#define BM 64
#define BN 128
#define BK 32

__device__ __forceinline__ unsigned pack_f16x2(float lo, float hi)
{
    _Float16 l = (_Float16)lo, h = (_Float16)hi;
    unsigned short ul = __builtin_bit_cast(unsigned short, l);
    unsigned short uh = __builtin_bit_cast(unsigned short, h);
    return ((unsigned)uh << 16) | ul;
}

// ---------------------------------------------------------------------------
// WMMA GEMM: C[M x N] = A[M x K](f32) * B[K x N](f32) (+ R)
// 64x128 block tile, BK=32, 128 threads = 4 waves; wave w owns rows
// [w*16,w*16+16) x 128 cols -> 8x v_wmma_f32_16x16x32_f16 per K-step.
// Double-buffered LDS; next tile's global loads are issued BEFORE the WMMA
// block and stored after it, so HBM latency hides under the matrix math.
// fp32 -> f16 conversion happens while staging, packed into b32 LDS stores
// laid out in the exact ISA 7.12.2 per-lane fragment order (each lane's
// fragment read is one contiguous 32B LDS load).
// ---------------------------------------------------------------------------
__global__ __launch_bounds__(128)
void gemm_wmma_f32(const float* __restrict__ A, int lda,
                   const float* __restrict__ B, int ldb,
                   const float* __restrict__ R,
                   float* __restrict__ C, int ldc,
                   int K)
{
    __shared__ __align__(32) unsigned AsU[2][BM * BK / 2];   // 2 x 4KB
    __shared__ __align__(32) unsigned BsU[2][BK * BN / 2];   // 2 x 8KB

    const int tid  = threadIdx.x;
    const int wave = tid >> 5;
    const int lane = tid & 31;
    const int m0   = blockIdx.y * BM;
    const int n0   = blockIdx.x * BN;
    const int nk   = K >> 5;

    v8f zero = {};
    v8f acc[8] = { zero, zero, zero, zero, zero, zero, zero, zero };

    v4f   areg[4];       // A staging: 4 quads of 4 consecutive k
    float breg[8][4];    // B staging: 8 quads of 4 consecutive k in one column

    // --- tile loaders (registers only, no waits forced before compute) ---
    auto loadA = [&](int k0) {
        #pragma unroll
        for (int q = 0; q < 4; ++q) {
            int idx = (q * 128 + tid) * 4;      // 0..2044 over 64x32 tile
            int m = idx >> 5, kk = idx & 31;
            areg[q] = *(const v4f*)(A + (size_t)(m0 + m) * lda + k0 + kk);
        }
    };
    auto loadB = [&](int k0) {
        #pragma unroll
        for (int q = 0; q < 8; ++q) {
            int qidx = q * 128 + tid;           // 0..1023 quads over 32x128
            int n = qidx & 127, kq = (qidx >> 7) << 2;
            #pragma unroll
            for (int i = 0; i < 4; ++i)
                breg[q][i] = B[(size_t)(k0 + kq + i) * ldb + n0 + n];
        }
    };
    // --- convert + swizzle + packed b32 LDS stores ---
    auto storeA = [&](int buf) {
        #pragma unroll
        for (int q = 0; q < 4; ++q) {
            int idx = (q * 128 + tid) * 4;
            int m = idx >> 5, kk = idx & 31, mblk = m >> 4;
            #pragma unroll
            for (int hp = 0; hp < 2; ++hp) {    // pairs (kk,kk+1),(kk+2,kk+3)
                int k = kk + hp * 2;
                int l = (m & 15) + (((k >> 3) & 1) << 4);
                int v = ((k & 7) >> 1) + ((k >> 4) << 2);
                AsU[buf][((mblk * 32 + l) << 3) + v] =
                    pack_f16x2(areg[q][hp * 2], areg[q][hp * 2 + 1]);
            }
        }
    };
    auto storeB = [&](int buf) {
        #pragma unroll
        for (int q = 0; q < 8; ++q) {
            int qidx = q * 128 + tid;
            int n = qidx & 127, kq = (qidx >> 7) << 2;
            int j = n >> 4;
            int l = (n & 15) + (((kq >> 3) & 1) << 4);  // bit3 const in quad
            int v = ((kq & 7) >> 1) + ((kq >> 4) << 2);
            BsU[buf][((j * 32 + l) << 3) + v]     = pack_f16x2(breg[q][0], breg[q][1]);
            BsU[buf][((j * 32 + l) << 3) + v + 1] = pack_f16x2(breg[q][2], breg[q][3]);
        }
    };

    // prologue: stage tile 0
    loadA(0); loadB(0);
    storeA(0); storeB(0);
    __syncthreads();

    int buf = 0;
    for (int kt = 0; kt < nk; ++kt) {
        const bool more = (kt + 1) < nk;
        if (more) {                      // issue next tile's loads (in flight
            loadA((kt + 1) << 5);        //  during the WMMA block below)
            loadB((kt + 1) << 5);
            __builtin_prefetch(B + (size_t)((kt + 2) << 5) * ldb + n0, 0, 0);
        }

        v16h a = *(const v16h*)&AsU[buf][(wave * 32 + lane) << 3];
        #pragma unroll
        for (int j = 0; j < 8; ++j) {
            v16h b = *(const v16h*)&BsU[buf][(j * 32 + lane) << 3];
            acc[j] = __builtin_amdgcn_wmma_f32_16x16x32_f16(
                false, a, false, b, (short)0, acc[j], false, false);
        }

        if (more) { storeA(buf ^ 1); storeB(buf ^ 1); }
        __syncthreads();
        buf ^= 1;
    }

    // epilogue: C VGPR r -> lanes 0-15: M=r, lanes 16-31: M=r+8 ; N = lane&15
    const int mbase = m0 + wave * 16 + ((lane >> 4) << 3);
    const int nbase = n0 + (lane & 15);
    #pragma unroll
    for (int j = 0; j < 8; ++j) {
        #pragma unroll
        for (int r = 0; r < 8; ++r) {
            size_t off = (size_t)(mbase + r) * ldc + nbase + j * 16;
            float val = acc[j][r];
            if (R != nullptr) val += R[off];
            C[off] = val;
        }
    }
}

// ---------------------------------------------------------------------------
// Elementwise / reduction helpers
// ---------------------------------------------------------------------------
__global__ __launch_bounds__(256)
void embed_kernel(const int* __restrict__ ids, const float* __restrict__ emb,
                  float* __restrict__ x)
{
    int t = blockIdx.x * 256 + threadIdx.x;          // 256*512
    int m = t >> 9, d = t & 511;
    x[t] = emb[(size_t)ids[m] * DIM + d];
}

__global__ __launch_bounds__(128)
void rmsnorm_kernel(const float* __restrict__ x, const float* __restrict__ w,
                    float* __restrict__ out)
{
    __shared__ float red[4];
    int row = blockIdx.x, tid = threadIdx.x;
    v4f v = *(const v4f*)(x + (size_t)row * DIM + tid * 4);
    float ss = v[0]*v[0] + v[1]*v[1] + v[2]*v[2] + v[3]*v[3];
    for (int o = 16; o > 0; o >>= 1) ss += __shfl_down(ss, o, 32);
    if ((tid & 31) == 0) red[tid >> 5] = ss;
    __syncthreads();
    float rs = rsqrtf((red[0] + red[1] + red[2] + red[3]) * (1.0f / DIM) + 1e-6f);
    v4f wv = *(const v4f*)(w + tid * 4);
    v4f o;
    o[0] = v[0]*rs*wv[0]; o[1] = v[1]*rs*wv[1];
    o[2] = v[2]*rs*wv[2]; o[3] = v[3]*rs*wv[3];
    *(v4f*)(out + (size_t)row * DIM + tid * 4) = o;
}

__global__ __launch_bounds__(128)
void halt_kernel(const float* __restrict__ h, const float* __restrict__ wh,
                 const float* __restrict__ bh, float* __restrict__ halt)
{
    __shared__ float red[4];
    int row = blockIdx.x, tid = threadIdx.x;
    v4f v = *(const v4f*)(h + (size_t)row * DIM + tid * 4);
    v4f w = *(const v4f*)(wh + tid * 4);
    float s = v[0]*w[0] + v[1]*w[1] + v[2]*w[2] + v[3]*w[3];
    for (int o = 16; o > 0; o >>= 1) s += __shfl_down(s, o, 32);
    if ((tid & 31) == 0) red[tid >> 5] = s;
    __syncthreads();
    if (tid == 0) {
        float z = red[0] + red[1] + red[2] + red[3] + bh[0];
        halt[row] = 1.0f / (1.0f + expf(-z));
    }
}

__global__ __launch_bounds__(256)
void rope_kernel(float* __restrict__ qkv)
{
    int t = blockIdx.x * 256 + threadIdx.x;          // 256*8*32
    int i = t & 31, h = (t >> 5) & 7, m = t >> 8;
    float inv = powf(10000.0f, -(2.0f * i) / (float)HD);
    float ang = m * inv, c = cosf(ang), s = sinf(ang);
    float* q = qkv + (size_t)m * 3 * DIM + h * HD;
    float* k = q + DIM;
    float q1 = q[i], q2 = q[i + 32];
    q[i] = q1 * c - q2 * s;  q[i + 32] = q1 * s + q2 * c;
    float k1 = k[i], k2 = k[i + 32];
    k[i] = k1 * c - k2 * s;  k[i + 32] = k1 * s + k2 * c;
}

__global__ __launch_bounds__(256)
void attn_scores_kernel(const float* __restrict__ qkv, float* __restrict__ attw)
{
    int q = blockIdx.x >> 3, h = blockIdx.x & 7;
    __shared__ float qv[HD];
    __shared__ float red[8];
    int k = threadIdx.x;
    if (k < HD) qv[k] = qkv[(size_t)q * 3 * DIM + h * HD + k];
    __syncthreads();
    const float* kr = qkv + (size_t)k * 3 * DIM + DIM + h * HD;
    float dot = 0.f;
    #pragma unroll
    for (int d = 0; d < HD; ++d) dot += qv[d] * kr[d];
    float s = (k <= q) ? dot * 0.125f : -1e30f;
    float mx = s;
    for (int o = 16; o > 0; o >>= 1) mx = fmaxf(mx, __shfl_down(mx, o, 32));
    if ((k & 31) == 0) red[k >> 5] = mx;
    __syncthreads();
    mx = red[0];
    #pragma unroll
    for (int i = 1; i < 8; ++i) mx = fmaxf(mx, red[i]);
    float e = __expf(s - mx);
    __syncthreads();
    float sm = e;
    for (int o = 16; o > 0; o >>= 1) sm += __shfl_down(sm, o, 32);
    if ((k & 31) == 0) red[k >> 5] = sm;
    __syncthreads();
    sm = red[0] + red[1] + red[2] + red[3] + red[4] + red[5] + red[6] + red[7];
    attw[((size_t)h * SEQ + q) * SEQ + k] = e / sm;
}

__global__ __launch_bounds__(256)
void attn_out_kernel(const float* __restrict__ qkv, const float* __restrict__ attw,
                     float* __restrict__ ao)
{
    int t = blockIdx.x * 256 + threadIdx.x;          // 256*512
    int m = t >> 9, c = t & 511, h = c >> 6, d = c & 63;
    const float* a = attw + ((size_t)h * SEQ + m) * SEQ;
    float acc = 0.f;
    for (int k = 0; k <= m; ++k)
        acc += a[k] * qkv[(size_t)k * 3 * DIM + 2 * DIM + h * HD + d];
    ao[(size_t)m * DIM + c] = acc;
}

__global__ __launch_bounds__(256)
void silu_mul_kernel(const float* __restrict__ g, const float* __restrict__ u,
                     float* __restrict__ act, int n)
{
    int t = blockIdx.x * 256 + threadIdx.x;
    if (t >= n) return;
    float x = g[t];
    act[t] = (x / (1.0f + expf(-x))) * u[t];
}

__global__ __launch_bounds__(256)
void gelu_kernel(float* __restrict__ g, int n)
{
    int t = blockIdx.x * 256 + threadIdx.x;
    if (t >= n) return;
    float x = g[t];
    float inner = 0.7978845608028654f * (x + 0.044715f * x * x * x);
    g[t] = 0.5f * x * (1.0f + tanhf(inner));
}

// ACT combine: halt probs -> per-step weights, ponder cost
__global__ __launch_bounds__(256)
void act_combine_kernel(const float* __restrict__ halt, float* __restrict__ ponder,
                        float* __restrict__ wout, float* __restrict__ wbuf)
{
    int m = threadIdx.x;                              // 256 tokens
    float p[NSTEP];
    p[0] = 0.f;                                       // MIN_STEPS = 1
    #pragma unroll
    for (int s = 1; s < NSTEP; ++s) p[s] = halt[s * SEQ + m];
    float cum = 0.f, rem = 0.f;
    int runcnt = 0;
    float w[NSTEP];
    #pragma unroll
    for (int s = 0; s < NSTEP; ++s) {
        float prev = cum;
        cum += p[s];
        bool running = prev < 0.99f;
        bool use_rem = running && ((cum >= 0.99f) || (s == NSTEP - 1));
        w[s] = running ? (use_rem ? 1.f - prev : p[s]) : 0.f;
        if (use_rem) rem += 1.f - prev;
        if (running) ++runcnt;
    }
    ponder[m] = (float)runcnt + rem;
    #pragma unroll
    for (int s = 0; s < NSTEP; ++s) { wout[m * NSTEP + s] = w[s]; wbuf[m * NSTEP + s] = w[s]; }
}

// a_cat[m, 0:512]    = sum_s w[m,s] * h_s[m,:]     (feeds w_y)
// a_cat[m, 512:1024] = sum_s w[m,s] * h_{s+1}[m,:] (feeds w_delta)
__global__ __launch_bounds__(256)
void acat_kernel(const float* __restrict__ hs, const float* __restrict__ wbuf,
                 float* __restrict__ acat)
{
    int t = blockIdx.x * 256 + threadIdx.x;          // 256*512
    int m = t >> 9, d = t & 511;
    float ay = 0.f, ad = 0.f;
    #pragma unroll
    for (int s = 0; s < NSTEP; ++s) {
        float w = wbuf[m * NSTEP + s];
        ay += w * hs[(size_t)s * SEQ * DIM + (size_t)m * DIM + d];
        ad += w * hs[(size_t)(s + 1) * SEQ * DIM + (size_t)m * DIM + d];
    }
    acat[(size_t)m * 1024 + d]       = ay;
    acat[(size_t)m * 1024 + 512 + d] = ad;
}

// ---------------------------------------------------------------------------
extern "C" void kernel_launch(void* const* d_in, const int* in_sizes, int n_in,
                              void* d_out, int out_size, void* d_ws, size_t ws_size,
                              hipStream_t stream)
{
    (void)in_sizes; (void)n_in; (void)out_size; (void)ws_size;

    const int*   ids     = (const int*)  d_in[0];
    const float* emb     = (const float*)d_in[1];
    const float* attn_nw = (const float*)d_in[2];
    const float* mlp_nw  = (const float*)d_in[3];
    const float* wqkv    = (const float*)d_in[4];
    const float* wo      = (const float*)d_in[5];
    const float* wg      = (const float*)d_in[6];
    const float* wu      = (const float*)d_in[7];
    const float* wd      = (const float*)d_in[8];
    const float* final_nw= (const float*)d_in[9];
    const float* rc_nw   = (const float*)d_in[10];
    const float* rc_w1   = (const float*)d_in[11];
    const float* rc_w2   = (const float*)d_in[12];
    const float* w_y     = (const float*)d_in[13];
    const float* w_delta = (const float*)d_in[14];
    const float* w_halt  = (const float*)d_in[15];
    const float* b_halt  = (const float*)d_in[16];
    float* out = (float*)d_out;

    // workspace layout (fp32)
    float* W = (float*)d_ws;
    float* x      = W;             W += SEQ * DIM;
    float* h      = W;             W += SEQ * DIM;
    float* qkv    = W;             W += SEQ * 3 * DIM;
    float* attw   = W;             W += NHEAD * SEQ * SEQ;
    float* ao     = W;             W += SEQ * DIM;
    float* gbuf   = W;             W += SEQ * RCF;       // max(FF,RCF)
    float* ubuf   = W;             W += SEQ * FF;
    float* act    = W;             W += SEQ * FF;
    float* states = W;             W += SEQ * DIM;
    float* hs     = W;             W += (NSTEP + 1) * SEQ * DIM;
    float* halt   = W;             W += NSTEP * SEQ;
    float* wbuf   = W;             W += SEQ * NSTEP;
    float* acat   = W;             W += SEQ * 1024;

    const dim3 blk128(128), blk256(256);

    embed_kernel<<<SEQ * DIM / 256, blk256, 0, stream>>>(ids, emb, x);

    // ---- backbone ----
    for (int l = 0; l < NL; ++l) {
        rmsnorm_kernel<<<SEQ, blk128, 0, stream>>>(x, attn_nw + l * DIM, h);
        gemm_wmma_f32<<<dim3(3 * DIM / BN, SEQ / BM), blk128, 0, stream>>>(
            h, DIM, wqkv + (size_t)l * DIM * 3 * DIM, 3 * DIM, nullptr, qkv, 3 * DIM, DIM);
        rope_kernel<<<SEQ * NHEAD * 32 / 256, blk256, 0, stream>>>(qkv);
        attn_scores_kernel<<<SEQ * NHEAD, blk256, 0, stream>>>(qkv, attw);
        attn_out_kernel<<<SEQ * DIM / 256, blk256, 0, stream>>>(qkv, attw, ao);
        gemm_wmma_f32<<<dim3(DIM / BN, SEQ / BM), blk128, 0, stream>>>(
            ao, DIM, wo + (size_t)l * DIM * DIM, DIM, x, x, DIM, DIM);
        rmsnorm_kernel<<<SEQ, blk128, 0, stream>>>(x, mlp_nw + l * DIM, h);
        gemm_wmma_f32<<<dim3(FF / BN, SEQ / BM), blk128, 0, stream>>>(
            h, DIM, wg + (size_t)l * DIM * FF, FF, nullptr, gbuf, FF, DIM);
        gemm_wmma_f32<<<dim3(FF / BN, SEQ / BM), blk128, 0, stream>>>(
            h, DIM, wu + (size_t)l * DIM * FF, FF, nullptr, ubuf, FF, DIM);
        silu_mul_kernel<<<SEQ * FF / 256, blk256, 0, stream>>>(gbuf, ubuf, act, SEQ * FF);
        gemm_wmma_f32<<<dim3(DIM / BN, SEQ / BM), blk128, 0, stream>>>(
            act, FF, wd + (size_t)l * FF * DIM, DIM, x, x, DIM, FF);
    }
    rmsnorm_kernel<<<SEQ, blk128, 0, stream>>>(x, final_nw, states);

    // ---- ACT refinement recursion (no vocab GEMMs here) ----
    for (int s = 0; s < NSTEP; ++s) {
        float* hcur = hs + (size_t)s * SEQ * DIM;
        rmsnorm_kernel<<<SEQ, blk128, 0, stream>>>(states, rc_nw, hcur);
        halt_kernel<<<SEQ, blk128, 0, stream>>>(hcur, w_halt, b_halt, halt + s * SEQ);
        gemm_wmma_f32<<<dim3(RCF / BN, SEQ / BM), blk128, 0, stream>>>(
            hcur, DIM, rc_w1, RCF, nullptr, gbuf, RCF, DIM);
        gelu_kernel<<<SEQ * RCF / 256, blk256, 0, stream>>>(gbuf, SEQ * RCF);
        gemm_wmma_f32<<<dim3(DIM / BN, SEQ / BM), blk128, 0, stream>>>(
            gbuf, RCF, rc_w2, DIM, states, states, DIM, RCF);
    }
    rmsnorm_kernel<<<SEQ, blk128, 0, stream>>>(states, rc_nw, hs + (size_t)NSTEP * SEQ * DIM);

    // ---- ACT combine: weights/ponder, then only TWO vocab GEMMs ----
    float* out_ponder  = out + (size_t)SEQ * VOC;
    float* out_weights = out_ponder + SEQ;
    act_combine_kernel<<<1, blk256, 0, stream>>>(halt, out_ponder, out_weights, wbuf);
    acat_kernel<<<SEQ * DIM / 256, blk256, 0, stream>>>(hs, wbuf, acat);

    gemm_wmma_f32<<<dim3(VOC / BN, SEQ / BM), blk128, 0, stream>>>(
        acat, 1024, w_y, VOC, nullptr, out, VOC, DIM);
    gemm_wmma_f32<<<dim3(VOC / BN, SEQ / BM), blk128, 0, stream>>>(
        acat + DIM, 1024, w_delta, VOC, out, out, VOC, DIM);
}
